// ShiftedWindowAttention_63625645523580
// MI455X (gfx1250) — matrix operand order repaired and checked by reading, not compiled, not run on hardware
//
#include <hip/hip_runtime.h>

typedef __attribute__((ext_vector_type(16))) _Float16 v16h;
typedef __attribute__((ext_vector_type(8)))  _Float16 v8h;
typedef __attribute__((ext_vector_type(8)))  float    v8f;

// ---------------------------------------------------------------------------
// WMMA fragment loaders (wave32, 16x16x32 f16).
// A 16x32 (MxK): lane<16 holds K {0..7,16..23}, lane>=16 holds K {8..15,24..31}
// B 32x16 (KxN): lane<16 holds K {0..7,16..23}, lane>=16 holds K {8..15,24..31}
// p points at the start of this fragment's 32 contiguous K values for this
// lane's row (A) / column (B). All chunks are 16-byte aligned.
// ---------------------------------------------------------------------------
__device__ __forceinline__ v16h load_a_frag(const _Float16* p, int hi) {
  const v8h c0 = *reinterpret_cast<const v8h*>(p + hi * 8);
  const v8h c1 = *reinterpret_cast<const v8h*>(p + 16 + hi * 8);
  v16h r;
#pragma unroll
  for (int e = 0; e < 8; ++e) { r[e] = c0[e]; r[e + 8] = c1[e]; }
  return r;
}

__device__ __forceinline__ v16h load_b_frag(const _Float16* p, int hi) {
  const v8h c0 = *reinterpret_cast<const v8h*>(p + hi * 8);
  const v8h c1 = *reinterpret_cast<const v8h*>(p + 16 + hi * 8);
  v16h r;
#pragma unroll
  for (int e = 0; e < 8; ++e) { r[e] = c0[e]; r[e + 8] = c1[e]; }
  return r;
}

#define WMMA_F16(a, b, c) \
  __builtin_amdgcn_wmma_f32_16x16x32_f16(false, (a), false, (b), (short)0, (c), false, false)

// ---------------------------------------------------------------------------
// Kernel 1: per-window QKV projection.  [64(tok) x 128] @ [128 x 384] -> f16
// qkv layout: [win][which(3)][head(4)][64][32]
// ---------------------------------------------------------------------------
__global__ void __launch_bounds__(256) qkv_win_kernel(
    const float* __restrict__ x, const float* __restrict__ w,
    const float* __restrict__ bias, _Float16* __restrict__ qkv) {
  const int win  = blockIdx.x;
  const int b    = win >> 8;
  const int wi   = win & 255;
  const int wh   = wi >> 4;
  const int ww   = wi & 15;
  const int tid  = threadIdx.x;
  const int lane = tid & 31;
  const int wave = tid >> 5;
  const int hi   = (lane >> 4) & 1;
  const int ln   = lane & 15;

  __shared__ _Float16 lx[64 * 136];   // x tile, f16, stride 136 halves
  __shared__ _Float16 lw[384 * 40];   // weight slab [n][k32], stride 40 halves

  // Gather shifted-window tokens (roll by -3,-3), convert f32->f16, pad to 64.
  for (int i = tid; i < 64 * 32; i += 256) {
    const int row = i >> 5;
    const int c4  = (i & 31) << 2;
    float4 v = make_float4(0.f, 0.f, 0.f, 0.f);
    if (row < 49) {
      const int ih = row / 7, iw = row % 7;
      const int sh = (wh * 7 + ih + 3) % 112;
      const int sw = (ww * 7 + iw + 3) % 112;
      v = *reinterpret_cast<const float4*>(
          x + (((size_t)b * 112 + sh) * 112 + sw) * 128 + c4);
    }
    _Float16* d = &lx[row * 136 + c4];
    d[0] = (_Float16)v.x; d[1] = (_Float16)v.y;
    d[2] = (_Float16)v.z; d[3] = (_Float16)v.w;
  }

  const v8f z = {0.f, 0.f, 0.f, 0.f, 0.f, 0.f, 0.f, 0.f};
  v8f acc[12];
#pragma unroll
  for (int t = 0; t < 12; ++t) acc[t] = z;

  for (int ks = 0; ks < 4; ++ks) {
    __syncthreads();  // previous slab consumed (and lx ready at ks==0)
    // Stream weight slab W[n][ks*32 .. +31] -> lw[n][k] (f16).
    for (int i = tid; i < 384 * 8; i += 256) {
      const int n  = i >> 3;
      const int k4 = (i & 7) << 2;
      const float4 v = *reinterpret_cast<const float4*>(w + n * 128 + ks * 32 + k4);
      _Float16* d = &lw[n * 40 + k4];
      d[0] = (_Float16)v.x; d[1] = (_Float16)v.y;
      d[2] = (_Float16)v.z; d[3] = (_Float16)v.w;
    }
    __syncthreads();
#pragma unroll
    for (int ti = 0; ti < 12; ++ti) {
      const int tile = wave + ti * 8;      // 96 tiles = 4 Mt x 24 Nt
      const int mt = tile & 3;
      const int nt = tile >> 2;
      const v16h a  = load_a_frag(&lx[(mt * 16 + ln) * 136 + ks * 32], hi);
      const v16h bb = load_b_frag(&lw[(nt * 16 + ln) * 40], hi);
      acc[ti] = WMMA_F16(a, bb, acc[ti]);
    }
  }

  // Epilogue: add bias, scale Q by 1/sqrt(32), store f16 window-major.
#pragma unroll
  for (int ti = 0; ti < 12; ++ti) {
    const int tile = wave + ti * 8;
    const int mt = tile & 3;
    const int nt = tile >> 2;
#pragma unroll
    for (int r = 0; r < 8; ++r) {
      const int m = mt * 16 + hi * 8 + r;
      const int n = nt * 16 + ln;
      float val = acc[ti][r] + bias[n];
      const int which = n >> 7;
      const int rem   = n & 127;
      const int h     = rem >> 5;
      const int d     = rem & 31;
      if (which == 0) val *= 0.17677669529663687f;  // HD^-0.5
      qkv[((((size_t)win * 3 + which) * 4 + h) * 64 + m) * 32 + d] = (_Float16)val;
    }
  }
}

// ---------------------------------------------------------------------------
// Kernel 2: attention. 1 wave = 1 (window, head). 128 threads = 4 heads.
// S = Q K^T (16 WMMA, frags straight from global), bias+mask+softmax,
// O = P V (16 WMMA via LDS-staged P and V^T).  O layout: [win][64][head*32+d].
// ---------------------------------------------------------------------------
__global__ void __launch_bounds__(128) attn_win_kernel(
    const _Float16* __restrict__ qkv, const float* __restrict__ btab,
    _Float16* __restrict__ obuf) {
  const int tid  = threadIdx.x;
  const int lane = tid & 31;
  const int wave = tid >> 5;
  const int hi   = (lane >> 4) & 1;
  const int ln   = lane & 15;
  const int win  = blockIdx.x;
  const int head = wave;
  const int wi   = win & 255;
  const int wh   = wi >> 4;
  const int ww   = wi & 15;

  __shared__ _Float16 sP[4][64 * 72];
  __shared__ _Float16 sVT[4][32 * 72];
  __shared__ float    lbias[676];      // 169 x 4

  for (int i = tid; i < 676; i += 128) lbias[i] = btab[i];

  const _Float16* qb = qkv + (((size_t)win * 3 + 0) * 4 + head) * 2048;
  const _Float16* kb = qkv + (((size_t)win * 3 + 1) * 4 + head) * 2048;
  const _Float16* vb = qkv + (((size_t)win * 3 + 2) * 4 + head) * 2048;
  _Float16* P  = sP[wave];
  _Float16* vT = sVT[wave];

  // Stage V transposed: vT[d][m].
  for (int i = lane; i < 64 * 32; i += 32) {
    const int m = i >> 5, d = i & 31;
    vT[d * 72 + m] = vb[i];
  }
  __syncthreads();  // lbias + vT visible

  // S = Q K^T : single K-step of 32 (full head dim).
  v16h af[4], bf[4];
#pragma unroll
  for (int mt = 0; mt < 4; ++mt) af[mt] = load_a_frag(qb + (mt * 16 + ln) * 32, hi);
#pragma unroll
  for (int nt = 0; nt < 4; ++nt) bf[nt] = load_b_frag(kb + (nt * 16 + ln) * 32, hi);

  const v8f z = {0.f, 0.f, 0.f, 0.f, 0.f, 0.f, 0.f, 0.f};
  v8f s[4][4];
#pragma unroll
  for (int mt = 0; mt < 4; ++mt)
#pragma unroll
    for (int nt = 0; nt < 4; ++nt)
      s[mt][nt] = WMMA_F16(af[mt], bf[nt], z);

  // Relative-position bias + shift mask + softmax (row lives in one half-wave).
#pragma unroll
  for (int mt = 0; mt < 4; ++mt) {
#pragma unroll
    for (int r = 0; r < 8; ++r) {
      const int m  = mt * 16 + hi * 8 + r;
      const int ih = m / 7, iw = m % 7;
      const int gh = wh * 7 + ih, gw = ww * 7 + iw;
      const int lh = (gh < 105) ? 0 : ((gh < 109) ? 1 : 2);
      const int lv = (gw < 105) ? 0 : ((gw < 109) ? 1 : 2);
      const int labm = lh * 3 + lv;
      float mx = -3.0e38f;
#pragma unroll
      for (int nt = 0; nt < 4; ++nt) {
        const int n = nt * 16 + ln;
        float sv = s[mt][nt][r];
        if (n >= 49) {
          sv = -1.0e30f;                       // mask pad columns
        } else if (m < 49) {
          const int jh = n / 7, jw = n % 7;
          const int idx = (ih - jh + 6) * 13 + (iw - jw + 6);
          sv += lbias[idx * 4 + head];
          const int gjh = wh * 7 + jh, gjw = ww * 7 + jw;
          const int ljh = (gjh < 105) ? 0 : ((gjh < 109) ? 1 : 2);
          const int ljw = (gjw < 105) ? 0 : ((gjw < 109) ? 1 : 2);
          if (labm != ljh * 3 + ljw) sv -= 100.0f;
        }
        s[mt][nt][r] = sv;
        mx = fmaxf(mx, sv);
      }
#pragma unroll
      for (int off = 1; off < 16; off <<= 1)
        mx = fmaxf(mx, __shfl_xor(mx, off, 32));
      float sum = 0.f;
#pragma unroll
      for (int nt = 0; nt < 4; ++nt) {
        const float e = __expf(s[mt][nt][r] - mx);
        s[mt][nt][r] = e;
        sum += e;
      }
#pragma unroll
      for (int off = 1; off < 16; off <<= 1)
        sum += __shfl_xor(sum, off, 32);
      const float inv = 1.f / sum;
#pragma unroll
      for (int nt = 0; nt < 4; ++nt)
        P[m * 72 + nt * 16 + ln] = (_Float16)(s[mt][nt][r] * inv);
    }
  }
  __syncthreads();

  // O = P V : K = 64 (2 k-steps), N = 32 (2 tiles).
  v8f o[4][2];
#pragma unroll
  for (int mt = 0; mt < 4; ++mt)
#pragma unroll
    for (int nt = 0; nt < 2; ++nt) o[mt][nt] = z;

#pragma unroll
  for (int ks = 0; ks < 2; ++ks) {
#pragma unroll
    for (int mt = 0; mt < 4; ++mt) {
      const v16h a = load_a_frag(&P[(mt * 16 + ln) * 72 + ks * 32], hi);
#pragma unroll
      for (int nt = 0; nt < 2; ++nt) {
        const v16h bb = load_b_frag(&vT[(nt * 16 + ln) * 72 + ks * 32], hi);
        o[mt][nt] = WMMA_F16(a, bb, o[mt][nt]);
      }
    }
  }

#pragma unroll
  for (int mt = 0; mt < 4; ++mt)
#pragma unroll
    for (int nt = 0; nt < 2; ++nt)
#pragma unroll
      for (int r = 0; r < 8; ++r) {
        const int m = mt * 16 + hi * 8 + r;
        const int d = nt * 16 + ln;
        obuf[((size_t)win * 64 + m) * 128 + head * 32 + d] = (_Float16)o[mt][nt][r];
      }
}

// ---------------------------------------------------------------------------
// Kernel 3: output projection + inverse shift scatter (f32 output).
// ---------------------------------------------------------------------------
__global__ void __launch_bounds__(256) proj_win_kernel(
    const _Float16* __restrict__ obuf, const float* __restrict__ w,
    const float* __restrict__ bias, float* __restrict__ out) {
  const int win  = blockIdx.x;
  const int b    = win >> 8;
  const int wi   = win & 255;
  const int wh   = wi >> 4;
  const int ww   = wi & 15;
  const int tid  = threadIdx.x;
  const int lane = tid & 31;
  const int wave = tid >> 5;
  const int hi   = (lane >> 4) & 1;
  const int ln   = lane & 15;

  __shared__ _Float16 lw[128 * 136];  // proj_w [c][k] f16, stride 136
  for (int i = tid; i < 128 * 32; i += 256) {
    const int c  = i >> 5;
    const int k4 = (i & 31) << 2;
    const float4 v = *reinterpret_cast<const float4*>(w + c * 128 + k4);
    _Float16* d = &lw[c * 136 + k4];
    d[0] = (_Float16)v.x; d[1] = (_Float16)v.y;
    d[2] = (_Float16)v.z; d[3] = (_Float16)v.w;
  }
  __syncthreads();

  const v8f z = {0.f, 0.f, 0.f, 0.f, 0.f, 0.f, 0.f, 0.f};
  v8f acc[4];
#pragma unroll
  for (int t = 0; t < 4; ++t) acc[t] = z;

  for (int ks = 0; ks < 4; ++ks) {
#pragma unroll
    for (int ti = 0; ti < 4; ++ti) {
      const int tile = wave + ti * 8;    // 32 tiles = 4 Mt x 8 Nt
      const int mt = tile & 3;
      const int nt = tile >> 2;
      const v16h a = load_a_frag(
          obuf + ((size_t)win * 64 + mt * 16 + ln) * 128 + ks * 32, hi);
      const v16h bb = load_b_frag(&lw[(nt * 16 + ln) * 136 + ks * 32], hi);
      acc[ti] = WMMA_F16(a, bb, acc[ti]);
    }
  }

#pragma unroll
  for (int ti = 0; ti < 4; ++ti) {
    const int tile = wave + ti * 8;
    const int mt = tile & 3;
    const int nt = tile >> 2;
#pragma unroll
    for (int r = 0; r < 8; ++r) {
      const int m = mt * 16 + hi * 8 + r;
      if (m < 49) {
        const int n  = nt * 16 + ln;
        const int ih = m / 7, iw = m % 7;
        const int fh = (wh * 7 + ih + 3) % 112;   // roll back by +3
        const int fw = (ww * 7 + iw + 3) % 112;
        out[(((size_t)b * 112 + fh) * 112 + fw) * 128 + n] = acc[ti][r] + bias[n];
      }
    }
  }
}

// ---------------------------------------------------------------------------
extern "C" void kernel_launch(void* const* d_in, const int* in_sizes, int n_in,
                              void* d_out, int out_size, void* d_ws, size_t ws_size,
                              hipStream_t stream) {
  (void)n_in; (void)out_size; (void)ws_size;
  const float* x          = (const float*)d_in[0];
  const float* qkv_w      = (const float*)d_in[1];
  const float* qkv_b      = (const float*)d_in[2];
  const float* proj_w     = (const float*)d_in[3];
  const float* proj_b     = (const float*)d_in[4];
  const float* bias_table = (const float*)d_in[5];

  const int Btot = in_sizes[0] / (112 * 112 * 128);   // 32
  const int nwin = Btot * 256;                        // 8192 windows

  _Float16* qkv  = (_Float16*)d_ws;                               // 384 MB
  _Float16* obuf = qkv + (size_t)nwin * 3 * 4 * 64 * 32;          // +134 MB

  qkv_win_kernel<<<nwin, 256, 0, stream>>>(x, qkv_w, qkv_b, qkv);
  attn_win_kernel<<<nwin, 128, 0, stream>>>(qkv, bias_table, obuf);
  proj_win_kernel<<<nwin, 256, 0, stream>>>(obuf, proj_w, proj_b, (float*)d_out);
}
